// WavLM_38955353375365
// MI455X (gfx1250) — matrix operand order repaired and checked by reading, not compiled, not run on hardware
//
#include <hip/hip_runtime.h>
#include <math.h>

typedef _Float16 half_t;
typedef __attribute__((ext_vector_type(8)))  _Float16 v8h;
typedef __attribute__((ext_vector_type(16))) _Float16 v16h;
typedef __attribute__((ext_vector_type(8)))  float    v8f;

#define B_  2
#define S_  1024
#define C_  768
#define H_  12
#define D_  64
#define FF_ 3072
#define L_  4

static __device__ __forceinline__ v16h cat8(v8h a, v8h b) {
  return __builtin_shufflevector(a, b, 0,1,2,3,4,5,6,7,8,9,10,11,12,13,14,15);
}

// A fragment (16 x 32 f16): row = lane%16; K chunks {ks..ks+7, ks+16..ks+23}, ks=(lane<16?0:8)
static __device__ __forceinline__ v16h frag_a(const half_t* base, int ld) {
  const int lane = threadIdx.x & 31;
  const half_t* p = base + (lane & 15) * ld + ((lane >> 4) << 3);
  return cat8(*(const v8h*)p, *(const v8h*)(p + 16));
}

// B fragment (32 x 16 f16, stored N-major rows of K): col = lane%16; K chunks {ks..ks+7, ks+16..ks+23}, ks=(lane<16?0:8)
static __device__ __forceinline__ v16h frag_b(const half_t* base, int ld) {
  const int lane = threadIdx.x & 31;
  const half_t* p = base + (lane & 15) * ld + ((lane >> 4) << 3);
  return cat8(*(const v8h*)p, *(const v8h*)(p + 16));
}

static __device__ __forceinline__ v8f wmma_f16(v16h a, v16h b, v8f c) {
  return __builtin_amdgcn_wmma_f32_16x16x32_f16(false, a, false, b, (short)0, c, false, false);
}

// CDNA5 async memory->LDS copy (ASYNCcnt-tracked, no VGPR round trip).
// lds_off = low 32 bits of a generic pointer to LDS = LDS byte address.
static __device__ __forceinline__ void async_ld16(unsigned lds_off, const void* gptr) {
  asm volatile("global_load_async_to_lds_b128 %0, %1, off"
               :: "v"(lds_off), "v"(gptr) : "memory");
}
static __device__ __forceinline__ void wait_async_le3() {
  asm volatile("s_wait_asynccnt 3" ::: "memory");
}
static __device__ __forceinline__ void wait_async_le0() {
  asm volatile("s_wait_asynccnt 0" ::: "memory");
}

// ---------------------------------------------------------------------------
// NT GEMM: C[M,N] = A[M,K] * B[N,K]^T (+bias).  Tile 128(M) x 64(N) x 32(K),
// 8 waves (wave = 32x32), double-buffered LDS filled by async-to-LDS copies.
// EPI: 0 -> f32 out (+bias); 1 -> f16 out (+bias)*scale; 2 -> f16 out gelu(+bias)
// ---------------------------------------------------------------------------
template<int EPI>
__global__ __launch_bounds__(256)
void gemm_nt(const half_t* __restrict__ A, const half_t* __restrict__ Bw,
             const float* __restrict__ bias, float* __restrict__ Cf,
             half_t* __restrict__ Ch, int M, int N, int K, float scale) {
  constexpr int LDT = 40;                       // 32 + 8 halfs pad (80B stride, 16B aligned)
  __shared__ __align__(16) half_t sA[2][128 * LDT];
  __shared__ __align__(16) half_t sB[2][64 * LDT];
  const int tid = threadIdx.x;
  const int m0 = blockIdx.y * 128, n0 = blockIdx.x * 64;
  const int wid = tid >> 5;
  const int wm = (wid >> 1) * 32;               // 0,32,64,96
  const int wn = (wid & 1) * 32;                // 0,32
  const int rA0 = tid >> 2, kcT = (tid & 3) << 3;  // staging chunk: row rA0, K offset kcT

  const half_t* gA0 = &A [(size_t)(m0 + rA0)      * K + kcT];
  const half_t* gA1 = &A [(size_t)(m0 + rA0 + 64) * K + kcT];
  const half_t* gB0 = &Bw[(size_t)(n0 + rA0)      * K + kcT];
  unsigned offA0[2], offA1[2], offB0[2];
#pragma unroll
  for (int d = 0; d < 2; ++d) {
    offA0[d] = (unsigned)(uintptr_t)&sA[d][rA0 * LDT + kcT];
    offA1[d] = (unsigned)(uintptr_t)&sA[d][(rA0 + 64) * LDT + kcT];
    offB0[d] = (unsigned)(uintptr_t)&sB[d][rA0 * LDT + kcT];
  }
  // prologue: fill buffer 0
  async_ld16(offA0[0], gA0);
  async_ld16(offA1[0], gA1);
  async_ld16(offB0[0], gB0);

  v8f acc[2][2] = {};
  int cur = 0;
  for (int kk = 0; kk < K; kk += 32, cur ^= 1) {
    const bool has_next = (kk + 32) < K;
    if (has_next) {                             // start next tile's copies first
      async_ld16(offA0[cur ^ 1], gA0 + kk + 32);
      async_ld16(offA1[cur ^ 1], gA1 + kk + 32);
      async_ld16(offB0[cur ^ 1], gB0 + kk + 32);
      wait_async_le3();                         // current tile (oldest 3) landed
    } else {
      wait_async_le0();
    }
    __syncthreads();                            // cross-wave visibility of LDS
    v16h af[2], bf[2];
#pragma unroll
    for (int i = 0; i < 2; ++i) af[i] = frag_a(&sA[cur][(wm + i * 16) * LDT], LDT);
#pragma unroll
    for (int j = 0; j < 2; ++j) bf[j] = frag_b(&sB[cur][(wn + j * 16) * LDT], LDT);
#pragma unroll
    for (int i = 0; i < 2; ++i)
#pragma unroll
      for (int j = 0; j < 2; ++j) acc[i][j] = wmma_f16(af[i], bf[j], acc[i][j]);
    __syncthreads();                            // done reading before buffer reuse
  }

  const int lane = tid & 31;
  const int nIn = lane & 15;
  const int mOff = (lane >> 4) << 3;
#pragma unroll
  for (int i = 0; i < 2; ++i) {
#pragma unroll
    for (int j = 0; j < 2; ++j) {
      int gn = n0 + wn + j * 16 + nIn;
      float bv = bias[gn];
#pragma unroll
      for (int r = 0; r < 8; ++r) {
        int gm = m0 + wm + i * 16 + mOff + r;
        float v = acc[i][j][r] + bv;
        if (EPI == 0) {
          Cf[(size_t)gm * N + gn] = v;
        } else if (EPI == 1) {
          Ch[(size_t)gm * N + gn] = (half_t)(v * scale);
        } else {
          Ch[(size_t)gm * N + gn] = (half_t)(0.5f * v * (1.0f + erff(v * 0.70710678118f)));
        }
      }
    }
  }
}

// ---------------------------------------------------------------------------
// Flash attention: block = (64 q rows, head hh, batch b); 4 waves, 16 rows/wave.
// q is pre-scaled by 1/sqrt(D).  scores += gate[b,q,h]*pos_bias[h,q,k] + keypad.
// K tile staged with async-to-LDS (overlapped with the manual V transpose).
// ---------------------------------------------------------------------------
__global__ __launch_bounds__(128)
void attn_kernel(const half_t* __restrict__ q, const half_t* __restrict__ k,
                 const half_t* __restrict__ v, const float* __restrict__ gate,
                 const float* __restrict__ pos_bias, const float* __restrict__ mask,
                 half_t* __restrict__ ao) {
  constexpr int LDK = 80, LDV = 80, LDP = 80;   // 160B strides (16B aligned)
  __shared__ __align__(16) half_t sK [64 * LDK];
  __shared__ __align__(16) half_t sVt[64 * LDV];
  __shared__ __align__(16) half_t sP [4 * 16 * LDP];
  const int b = blockIdx.z, hh = blockIdx.y;
  const int q0 = blockIdx.x * 64;
  const int tid = threadIdx.x;
  const int w = tid >> 5, lane = tid & 31;
  const int nIn = lane & 15, mOff = (lane >> 4) << 3;
  half_t* sPw = &sP[w * 16 * LDP];

  // Q fragments: rows q0+w*16 .. +15, K-dim = D = 64 (two 16x32 A fragments)
  v16h aq0, aq1;
  {
    int row = q0 + w * 16 + (lane & 15);
    const half_t* p = q + ((size_t)(b * S_) + row) * C_ + hh * D_ + ((lane >> 4) << 3);
    aq0 = cat8(*(const v8h*)p,        *(const v8h*)(p + 16));
    aq1 = cat8(*(const v8h*)(p + 32), *(const v8h*)(p + 48));
  }
  float g8[8];
#pragma unroll
  for (int r = 0; r < 8; ++r) {
    int row = q0 + w * 16 + mOff + r;
    g8[r] = gate[((size_t)b * S_ + row) * H_ + hh];
  }
  float mI[8], lI[8];
#pragma unroll
  for (int r = 0; r < 8; ++r) { mI[r] = -1e30f; lI[r] = 0.0f; }
  v8f o[4] = {};

  const half_t* kbase = k + (size_t)(b * S_) * C_ + hh * D_;
  const half_t* vbase = v + (size_t)(b * S_) * C_ + hh * D_;

  for (int kt = 0; kt < S_; kt += 64) {
    __syncthreads();                            // prior sK/sVt readers done
    // async-copy the K tile (contiguous) while we transpose V manually
#pragma unroll
    for (int t = 0; t < 4; ++t) {               // 512 chunks of 8 halfs
      int c = tid + t * 128;
      int kr = c >> 3, dc = (c & 7) << 3;
      async_ld16((unsigned)(uintptr_t)&sK[kr * LDK + dc],
                 &kbase[(size_t)(kt + kr) * C_ + dc]);
    }
#pragma unroll
    for (int t = 0; t < 4; ++t) {
      int c = tid + t * 128;
      int kr = c >> 3, dc = (c & 7) << 3;
      v8h vv = *(const v8h*)&vbase[(size_t)(kt + kr) * C_ + dc];
#pragma unroll
      for (int e = 0; e < 8; ++e) sVt[(dc + e) * LDV + kr] = vv[e];  // transpose V
    }
    wait_async_le0();
    __syncthreads();

    // 4 score tiles of 16 keys: S = Q * K^T  (K-dim = D, split 0..31 / 32..63)
    v8f sv[4];
#pragma unroll
    for (int nt = 0; nt < 4; ++nt) {
      v16h b0 = frag_b(&sK[(nt * 16) * LDK], LDK);
      v16h b1 = frag_b(&sK[(nt * 16) * LDK + 32], LDK);
      v8f s = {};
      s = wmma_f16(aq0, b0, s);
      s = wmma_f16(aq1, b1, s);
      int kc = kt + nt * 16 + nIn;
      float kp = (1.0f - mask[(size_t)b * S_ + kc]) * -10000.0f;
#pragma unroll
      for (int r = 0; r < 8; ++r) {
        int qrow = q0 + w * 16 + mOff + r;
        float pb = pos_bias[((size_t)hh * S_ + qrow) * S_ + kc];
        s[r] = s[r] + g8[r] * pb + kp;
      }
      sv[nt] = s;
    }

    // online softmax across this 64-key block (row = 16 lanes of a half-group)
#pragma unroll
    for (int r = 0; r < 8; ++r) {
      float tm = fmaxf(fmaxf(sv[0][r], sv[1][r]), fmaxf(sv[2][r], sv[3][r]));
#pragma unroll
      for (int sft = 1; sft < 16; sft <<= 1) tm = fmaxf(tm, __shfl_xor(tm, sft, 32));
      float mN = fmaxf(mI[r], tm);
      float cf = __expf(mI[r] - mN);
      float ps = 0.0f;
#pragma unroll
      for (int nt = 0; nt < 4; ++nt) {
        float p = __expf(sv[nt][r] - mN);
        sPw[(mOff + r) * LDP + nt * 16 + nIn] = (half_t)p;
        ps += p;
      }
#pragma unroll
      for (int sft = 1; sft < 16; sft <<= 1) ps += __shfl_xor(ps, sft, 32);
      lI[r] = lI[r] * cf + ps;
      mI[r] = mN;
#pragma unroll
      for (int f = 0; f < 4; ++f) o[f][r] *= cf;
    }

    // O += P * V   (A = probs 16x64, B = Vt: keys contiguous per d-column)
#pragma unroll
    for (int kc = 0; kc < 64; kc += 32) {
      v16h ap = frag_a(&sPw[kc], LDP);
#pragma unroll
      for (int f = 0; f < 4; ++f) {
        v16h bv = frag_b(&sVt[(f * 16) * LDV + kc], LDV);
        o[f] = wmma_f16(ap, bv, o[f]);
      }
    }
  }

#pragma unroll
  for (int f = 0; f < 4; ++f) {
#pragma unroll
    for (int r = 0; r < 8; ++r) {
      int qrow = q0 + w * 16 + mOff + r;
      float ov = o[f][r] / lI[r];
      ao[((size_t)(b * S_) + qrow) * C_ + hh * D_ + f * 16 + nIn] = (half_t)ov;
    }
  }
}

// ---------------------------------------------------------------------------
// LayerNorm kernels (row = b*S+s, 256 threads, 3 elems/thread)
// ---------------------------------------------------------------------------
__global__ __launch_bounds__(256)
void ln0_kernel(const float* __restrict__ x, const float* __restrict__ pos,
                const float* __restrict__ mask, const float* __restrict__ g,
                const float* __restrict__ bb, float* __restrict__ hf,
                half_t* __restrict__ hh) {
  __shared__ float rs[8], rq[8];
  const int row = blockIdx.x;
  const int tid = threadIdx.x;
  const float mk = mask[row];
  float vals[3], s = 0.0f, sq = 0.0f;
#pragma unroll
  for (int t = 0; t < 3; ++t) {
    int i = tid + t * 256;
    float hv = x[(size_t)row * C_ + i] * mk + pos[(size_t)row * C_ + i];
    vals[t] = hv; s += hv; sq += hv * hv;
  }
#pragma unroll
  for (int sh = 1; sh < 32; sh <<= 1) { s += __shfl_xor(s, sh, 32); sq += __shfl_xor(sq, sh, 32); }
  if ((tid & 31) == 0) { rs[tid >> 5] = s; rq[tid >> 5] = sq; }
  __syncthreads();
  float ts = 0.0f, tq = 0.0f;
#pragma unroll
  for (int i = 0; i < 8; ++i) { ts += rs[i]; tq += rq[i]; }
  const float mean = ts * (1.0f / C_);
  const float var  = tq * (1.0f / C_) - mean * mean;
  const float inv  = rsqrtf(var + 1e-5f);
#pragma unroll
  for (int t = 0; t < 3; ++t) {
    int i = tid + t * 256;
    float y = (vals[t] - mean) * inv * g[i] + bb[i];
    hf[(size_t)row * C_ + i] = y;
    hh[(size_t)row * C_ + i] = (half_t)y;
  }
}

__global__ __launch_bounds__(256)
void ln_res_kernel(const float* __restrict__ res, const float* __restrict__ del,
                   const float* __restrict__ g, const float* __restrict__ bb,
                   float* __restrict__ outf, half_t* __restrict__ outh) {
  __shared__ float rs[8], rq[8];
  const int row = blockIdx.x;
  const int tid = threadIdx.x;
  float vals[3], s = 0.0f, sq = 0.0f;
#pragma unroll
  for (int t = 0; t < 3; ++t) {
    int i = tid + t * 256;
    float hv = res[(size_t)row * C_ + i] + del[(size_t)row * C_ + i];
    vals[t] = hv; s += hv; sq += hv * hv;
  }
#pragma unroll
  for (int sh = 1; sh < 32; sh <<= 1) { s += __shfl_xor(s, sh, 32); sq += __shfl_xor(sq, sh, 32); }
  if ((tid & 31) == 0) { rs[tid >> 5] = s; rq[tid >> 5] = sq; }
  __syncthreads();
  float ts = 0.0f, tq = 0.0f;
#pragma unroll
  for (int i = 0; i < 8; ++i) { ts += rs[i]; tq += rq[i]; }
  const float mean = ts * (1.0f / C_);
  const float var  = tq * (1.0f / C_) - mean * mean;
  const float inv  = rsqrtf(var + 1e-5f);
#pragma unroll
  for (int t = 0; t < 3; ++t) {
    int i = tid + t * 256;
    float y = (vals[t] - mean) * inv * g[i] + bb[i];
    outf[(size_t)row * C_ + i] = y;
    outh[(size_t)row * C_ + i] = (half_t)y;
  }
}

// gate[b,s,h] = sigmoid(sum g[0:4]) * (sigmoid(sum g[4:8]) * gc[h] - 1) + 2
__global__ void gate_kernel(const float* __restrict__ hf, const float* __restrict__ gw,
                            const float* __restrict__ gb, const float* __restrict__ gc,
                            float* __restrict__ gate) {
  int idx = blockIdx.x * blockDim.x + threadIdx.x;
  if (idx >= B_ * S_ * H_) return;
  int hd = idx % H_;
  int bs = idx / H_;
  const float* hp = hf + (size_t)bs * C_ + hd * D_;
  float acc[8];
#pragma unroll
  for (int e = 0; e < 8; ++e) acc[e] = gb[e];
  for (int d = 0; d < D_; ++d) {
    float hv = hp[d];
#pragma unroll
    for (int e = 0; e < 8; ++e) acc[e] += hv * gw[e * D_ + d];
  }
  float sa = acc[0] + acc[1] + acc[2] + acc[3];
  float sb = acc[4] + acc[5] + acc[6] + acc[7];
  float ga  = 1.0f / (1.0f + __expf(-sa));
  float gbv = 1.0f / (1.0f + __expf(-sb));
  gate[idx] = ga * (gbv * gc[hd] - 1.0f) + 2.0f;
}

__global__ void cvt_f16_kernel(const float* __restrict__ s, half_t* __restrict__ d, int n) {
  int i = blockIdx.x * blockDim.x + threadIdx.x;
  if (i < n) d[i] = (half_t)s[i];
}

// ---------------------------------------------------------------------------
extern "C" void kernel_launch(void* const* d_in, const int* in_sizes, int n_in,
                              void* d_out, int out_size, void* d_ws, size_t ws_size,
                              hipStream_t stream) {
  (void)in_sizes; (void)n_in; (void)out_size; (void)ws_size;
  const float* x        = (const float*)d_in[0];
  const float* pos_emb  = (const float*)d_in[1];
  const float* attn_m   = (const float*)d_in[2];
  const float* pos_bias = (const float*)d_in[3];
  const float* ln0_g    = (const float*)d_in[4];
  const float* ln0_b    = (const float*)d_in[5];
  const float* Wq       = (const float*)d_in[6];
  const float* bq       = (const float*)d_in[7];
  const float* Wk       = (const float*)d_in[8];
  const float* bk       = (const float*)d_in[9];
  const float* Wv       = (const float*)d_in[10];
  const float* bv       = (const float*)d_in[11];
  const float* Wo       = (const float*)d_in[12];
  const float* bo       = (const float*)d_in[13];
  const float* gruW     = (const float*)d_in[14];
  const float* grub     = (const float*)d_in[15];
  const float* gruc     = (const float*)d_in[16];
  const float* ln1_g    = (const float*)d_in[17];
  const float* ln1_b    = (const float*)d_in[18];
  const float* W1       = (const float*)d_in[19];
  const float* b1       = (const float*)d_in[20];
  const float* W2       = (const float*)d_in[21];
  const float* b2       = (const float*)d_in[22];
  const float* ln2_g    = (const float*)d_in[23];
  const float* ln2_b    = (const float*)d_in[24];

  char* ws = (char*)d_ws;
  size_t off = 0;
  auto alloc = [&](size_t bytes) -> void* {
    void* p = (void*)(ws + off);
    off += (bytes + 255) & ~(size_t)255;
    return p;
  };
  const size_t NW  = (size_t)L_ * C_ * C_;
  const size_t NWF = (size_t)L_ * FF_ * C_;
  const size_t NA  = (size_t)B_ * S_ * C_;
  half_t* wq16 = (half_t*)alloc(NW * 2);
  half_t* wk16 = (half_t*)alloc(NW * 2);
  half_t* wv16 = (half_t*)alloc(NW * 2);
  half_t* wo16 = (half_t*)alloc(NW * 2);
  half_t* w116 = (half_t*)alloc(NWF * 2);
  half_t* w216 = (half_t*)alloc(NWF * 2);
  float*  hf   = (float*)alloc(NA * 4);
  half_t* hh   = (half_t*)alloc(NA * 2);
  float*  h1f  = (float*)alloc(NA * 4);
  half_t* h1h  = (half_t*)alloc(NA * 2);
  half_t* qb   = (half_t*)alloc(NA * 2);
  half_t* kb   = (half_t*)alloc(NA * 2);
  half_t* vb   = (half_t*)alloc(NA * 2);
  half_t* aoh  = (half_t*)alloc(NA * 2);
  float*  of32 = (float*)alloc(NA * 4);
  half_t* ff16 = (half_t*)alloc((size_t)B_ * S_ * FF_ * 2);
  float*  f2   = (float*)alloc(NA * 4);
  float*  gtb  = (float*)alloc((size_t)B_ * S_ * H_ * 4);

  // weights -> f16 (once per launch, deterministic)
  cvt_f16_kernel<<<(int)((NW  + 255) / 256), 256, 0, stream>>>(Wq, wq16, (int)NW);
  cvt_f16_kernel<<<(int)((NW  + 255) / 256), 256, 0, stream>>>(Wk, wk16, (int)NW);
  cvt_f16_kernel<<<(int)((NW  + 255) / 256), 256, 0, stream>>>(Wv, wv16, (int)NW);
  cvt_f16_kernel<<<(int)((NW  + 255) / 256), 256, 0, stream>>>(Wo, wo16, (int)NW);
  cvt_f16_kernel<<<(int)((NWF + 255) / 256), 256, 0, stream>>>(W1, w116, (int)NWF);
  cvt_f16_kernel<<<(int)((NWF + 255) / 256), 256, 0, stream>>>(W2, w216, (int)NWF);

  // h = LN0(x*mask + pos_emb)
  ln0_kernel<<<B_ * S_, 256, 0, stream>>>(x, pos_emb, attn_m, ln0_g, ln0_b, hf, hh);

  const dim3 g768(C_ / 64, (B_ * S_) / 128);    // (12,16) = 192 blocks
  const dim3 gff(FF_ / 64, (B_ * S_) / 128);    // (48,16) = 768 blocks
  const dim3 gattn(S_ / 64, H_, B_);            // (16,12,2)

  for (int l = 0; l < L_; ++l) {
    const half_t* WQl = wq16 + (size_t)l * C_ * C_;
    const half_t* WKl = wk16 + (size_t)l * C_ * C_;
    const half_t* WVl = wv16 + (size_t)l * C_ * C_;
    const half_t* WOl = wo16 + (size_t)l * C_ * C_;
    const half_t* W1l = w116 + (size_t)l * FF_ * C_;
    const half_t* W2l = w216 + (size_t)l * C_ * FF_;

    gemm_nt<1><<<g768, 256, 0, stream>>>(hh, WQl, bq + l * C_, nullptr, qb,
                                         B_ * S_, C_, C_, 0.125f);
    gemm_nt<1><<<g768, 256, 0, stream>>>(hh, WKl, bk + l * C_, nullptr, kb,
                                         B_ * S_, C_, C_, 1.0f);
    gemm_nt<1><<<g768, 256, 0, stream>>>(hh, WVl, bv + l * C_, nullptr, vb,
                                         B_ * S_, C_, C_, 1.0f);
    gate_kernel<<<(B_ * S_ * H_ + 255) / 256, 256, 0, stream>>>(
        hf, gruW + (size_t)l * 8 * D_, grub + (size_t)l * 8, gruc + (size_t)l * H_, gtb);
    attn_kernel<<<gattn, 128, 0, stream>>>(qb, kb, vb, gtb, pos_bias, attn_m, aoh);
    gemm_nt<0><<<g768, 256, 0, stream>>>(aoh, WOl, bo + l * C_, of32, nullptr,
                                         B_ * S_, C_, C_, 1.0f);
    ln_res_kernel<<<B_ * S_, 256, 0, stream>>>(hf, of32, ln1_g + l * C_, ln1_b + l * C_,
                                               h1f, h1h);
    gemm_nt<2><<<gff, 256, 0, stream>>>(h1h, W1l, b1 + l * FF_, nullptr, ff16,
                                        B_ * S_, FF_, C_, 1.0f);
    gemm_nt<0><<<g768, 256, 0, stream>>>(ff16, W2l, b2 + l * C_, f2, nullptr,
                                         B_ * S_, C_, FF_, 1.0f);
    ln_res_kernel<<<B_ * S_, 256, 0, stream>>>(h1f, f2, ln2_g + l * C_, ln2_b + l * C_,
                                               hf, hh);
  }

  hipMemcpyAsync(d_out, hf, NA * sizeof(float), hipMemcpyDeviceToDevice, stream);
}
